// ImuModelConvLstm_62208306315604
// MI455X (gfx1250) — matrix operand order repaired and checked by reading, not compiled, not run on hardware
//
#include <hip/hip_runtime.h>
#include <hip/hip_bf16.h>
#include <cmath>

// ---------------------------------------------------------------------------
// ConvLSTM x2 + BN + dense/softmax/global-mean for MI455X (gfx1250, wave32).
//
// Every 3x3 conv pair is a split-K implicit GEMM M=8192 (B*H*W), N=256 (4F),
// K = Kx + Kh, executed with V_WMMA_F32_16X16X32_F16 (f16 in, f32 acc).
// Weights pre-transposed to [Cout][K] f16; activations stored f16 NHWC so
// im2col A-fragments are contiguous b128 loads. B tiles are staged through a
// DOUBLE-BUFFERED 2x16KB LDS region (one barrier per k-step; staging of chunk
// s+1 overlaps compute on chunk s). sched_barriers pin the hot loop to:
//   16x ds_load_b128 (clause) -> one s_wait_dscnt -> 8 back-to-back WMMAs,
// trading ~64 VGPRs of B-fragment storage for a fully busy matrix pipe.
// Recurrence is serialized over T=16 via stream order.
// ---------------------------------------------------------------------------

typedef _Float16 v16h __attribute__((ext_vector_type(16)));
typedef _Float16 v8h  __attribute__((ext_vector_type(8)));
typedef float    v8f  __attribute__((ext_vector_type(8)));

#if defined(__has_builtin)
# if __has_builtin(__builtin_amdgcn_sched_barrier)
#  define SCHED_FENCE() __builtin_amdgcn_sched_barrier(0)
# endif
#endif
#ifndef SCHED_FENCE
# define SCHED_FENCE()
#endif

#define NB   8
#define NT   16
#define NH   32
#define NW   32
#define NCIN 6
#define NF   64
#define NCLS 10
#define MROWS (NB*NH*NW)      /* 8192 rows per timestep */
#define NCOLS (4*NF)          /* 256  output channels   */
#define KH    (9*NF)          /* 576  im2col K for h / bn(h) */
#define KX0   64              /* 54 real + pad          */
#define LDSB_HALVES (256*32)  /* one B stage buffer: 16 KB */

// ---------------------------------------------------------------------------
// Weight / BN prep: transpose to [Cout][K] f16, build BN0 scale/shift and
// BN1-folded dense weights.
// ---------------------------------------------------------------------------
__global__ void prep_kernel(const float* __restrict__ l0Wx, const float* __restrict__ l0Wh,
                            const float* __restrict__ l1Wx, const float* __restrict__ l1Wh,
                            const float* __restrict__ g0, const float* __restrict__ be0,
                            const float* __restrict__ mu0, const float* __restrict__ va0,
                            const float* __restrict__ g1, const float* __restrict__ be1,
                            const float* __restrict__ mu1, const float* __restrict__ va1,
                            const float* __restrict__ dW, const float* __restrict__ db,
                            _Float16* __restrict__ Wx0t, _Float16* __restrict__ Wh0t,
                            _Float16* __restrict__ Wx1t, _Float16* __restrict__ Wh1t,
                            float* __restrict__ scale0, float* __restrict__ shift0,
                            float* __restrict__ Wp, float* __restrict__ bp) {
  const int idx = blockIdx.x * 256 + threadIdx.x;     // grid covers 256*576
  if (idx < NCOLS * KH) {
    const int co = idx / KH, k = idx % KH;            // layout [co][k]
    Wh0t[idx] = (_Float16)l0Wh[k * NCOLS + co];       // src [tap*64+ci][co]
    Wx1t[idx] = (_Float16)l1Wx[k * NCOLS + co];
    Wh1t[idx] = (_Float16)l1Wh[k * NCOLS + co];
  }
  if (idx < NCOLS * KX0) {
    const int co = idx >> 6, k = idx & 63;            // k = tap*6+ci, pad >=54
    Wx0t[idx] = (k < 54) ? (_Float16)l0Wx[k * NCOLS + co] : (_Float16)0.f;
  }
  if (idx < NF) {
    const float s = g0[idx] * rsqrtf(va0[idx] + 1e-3f);
    scale0[idx] = s;
    shift0[idx] = be0[idx] - mu0[idx] * s;
  }
  if (idx < NF * NCLS) {                              // BN1 folded into dense
    const int f = idx / NCLS;
    const float s1 = g1[f] * rsqrtf(va1[f] + 1e-3f);
    Wp[idx] = dW[idx] * s1;
  }
  if (idx < NCLS) {
    float acc = db[idx];
    for (int f = 0; f < NF; ++f) {
      const float s1 = g1[f] * rsqrtf(va1[f] + 1e-3f);
      acc += (be1[f] - mu1[f] * s1) * dW[f * NCLS + idx];
    }
    bp[idx] = acc;
  }
}

// ---------------------------------------------------------------------------
// im2col of the raw input into f16: X0col[t][m=(b,y,x)][64] (54 real + pad).
// ---------------------------------------------------------------------------
__global__ void pack_x0_kernel(const float* __restrict__ inp, _Float16* __restrict__ X0col) {
  const int idx = blockIdx.x * 256 + threadIdx.x;     // 16*8192 rows
  const int t = idx >> 13, b = (idx >> 10) & 7, y = (idx >> 5) & 31, x = idx & 31;
  _Float16* row = X0col + (size_t)idx * KX0;
  int k = 0;
  for (int tap = 0; tap < 9; ++tap) {
    const int yy = y + tap / 3 - 1, xx = x + tap % 3 - 1;
    const bool ok = ((unsigned)yy < (unsigned)NH) && ((unsigned)xx < (unsigned)NW);
    const float* src = inp + ((((size_t)(b * NT + t) * NH + yy) * NW + xx) * NCIN);
    for (int ci = 0; ci < NCIN; ++ci, ++k)
      row[k] = ok ? (_Float16)src[ci] : (_Float16)0.f;
  }
  for (; k < KX0; ++k) row[k] = (_Float16)0.f;
}

// ---------------------------------------------------------------------------
// Split-K WMMA GEMM:  Z[8192][256] = A0 * B0 + A1 * B1
// Block = 256 thr (8 waves): 64x256 tile; wave = 16x128 = 8 WMMA tiles.
// ---------------------------------------------------------------------------
__device__ __forceinline__ void stageB(const _Float16* __restrict__ Bt, int Klen, int kk,
                                       int tid, _Float16* __restrict__ dst) {
  // thread tid owns output column `tid`: copy Bt[tid][kk..kk+32) -> LDS
  const _Float16* src = Bt + (size_t)tid * Klen + kk;
  *(v8h*)(dst +  0) = *(const v8h*)(src +  0);
  *(v8h*)(dst +  8) = *(const v8h*)(src +  8);
  *(v8h*)(dst + 16) = *(const v8h*)(src + 16);
  *(v8h*)(dst + 24) = *(const v8h*)(src + 24);
  if (kk + 64 < Klen) __builtin_prefetch(src + 64, 0, 1);   // global_prefetch_b8
}

__device__ __forceinline__ void mma_source(const _Float16* __restrict__ A, int Klen,
                                           int im2col, const _Float16* __restrict__ Bt,
                                           _Float16* ldsB, int tid, int l15, int hi,
                                           int b, int y, int x, int m, int nbase,
                                           v8f acc[8]) {
  const int nsteps = Klen >> 5;
  // prologue: stage chunk 0 into buffer 0
  stageB(Bt, Klen, 0, tid, ldsB + tid * 32);
  __syncthreads();

  for (int s = 0; s < nsteps; ++s) {
    const int kk = s << 5;
    _Float16* curbuf = ldsB + (s & 1) * LDSB_HALVES;
    // stage next chunk into the other buffer while computing this one
    if (s + 1 < nsteps)
      stageB(Bt, Klen, kk + 32, tid,
             ldsB + ((s + 1) & 1) * LDSB_HALVES + tid * 32);

    // A fragment: lane holds row m; halves = K {kk+hi*8..} U {kk+16+hi*8..}
    v8h lo = {}, hic = {};
    if (im2col) {
      const int tap = kk >> 6;                 // 32-step stays inside one tap
      const int cib = kk & 63;
      const int yy = y + tap / 3 - 1, xx = x + tap % 3 - 1;
      if ((unsigned)yy < (unsigned)NH && (unsigned)xx < (unsigned)NW) {
        const _Float16* p = A + ((((size_t)(b << 5) + yy) << 5) + xx) * NF + cib + hi * 8;
        lo  = *(const v8h*)p;
        hic = *(const v8h*)(p + 16);
      }
    } else {
      const _Float16* p = A + (size_t)m * Klen + kk + hi * 8;
      lo  = *(const v8h*)p;
      hic = *(const v8h*)(p + 16);
    }
    v16h afrag;
    #pragma unroll
    for (int j = 0; j < 8; ++j) { afrag[j] = lo[j]; afrag[j + 8] = hic[j]; }

    // Preload ALL 8 B fragments; sched fences keep the ds_load clause and the
    // WMMA chain from being re-interleaved by the scheduler.
    v16h bf[8];
    #pragma unroll
    for (int j = 0; j < 8; ++j) {            // lane holds column nbase+16j+l15
      const _Float16* bptr = curbuf + (nbase + j * 16 + l15) * 32 + hi * 8;
      const v8h b0 = *(const v8h*)bptr;      // K = hi*8 + 0..7
      const v8h b1 = *(const v8h*)(bptr + 16);// K = 16 + hi*8 + 0..7
      #pragma unroll
      for (int j2 = 0; j2 < 8; ++j2) { bf[j][j2] = b0[j2]; bf[j][j2 + 8] = b1[j2]; }
    }
    SCHED_FENCE();                           // all 16 ds_loads issued above
    #pragma unroll
    for (int j = 0; j < 8; ++j)
      acc[j] = __builtin_amdgcn_wmma_f32_16x16x32_f16(
          false, afrag, false, bf[j], (short)0, acc[j], false, false);
    SCHED_FENCE();                           // 8 back-to-back WMMAs above

    __syncthreads();   // reads of curbuf done; staging of next buffer visible
  }
}

__global__ __launch_bounds__(256) void gemm_conv_kernel(
    const _Float16* __restrict__ A0, int K0, int mode0, const _Float16* __restrict__ B0,
    const _Float16* __restrict__ A1, int K1, const _Float16* __restrict__ B1,
    float* __restrict__ Z) {
  __shared__ _Float16 ldsB[2 * LDSB_HALVES];           // 32 KB, double buffered
  const int tid  = threadIdx.x;
  const int lane = tid & 31;
  const int wave = tid >> 5;
  const int l15  = lane & 15;
  const int hi   = lane >> 4;
  const int mw   = wave & 3;                           // 4 M sub-tiles
  const int nw   = wave >> 2;                          // 2 N halves
  const int mbase = blockIdx.x * 64 + mw * 16;
  const int nbase = nw * 128;
  const int m = mbase + l15;                           // this lane's A row
  const int b = m >> 10, y = (m >> 5) & 31, x = m & 31;

  v8f acc[8];
  #pragma unroll
  for (int j = 0; j < 8; ++j) acc[j] = {};

  mma_source(A0, K0, mode0, B0, ldsB, tid, l15, hi, b, y, x, m, nbase, acc);
  __syncthreads();
  mma_source(A1, K1, 1,     B1, ldsB, tid, l15, hi, b, y, x, m, nbase, acc);

  // C/D layout: VGPR v -> M = mbase + v + 8*hi ; lane -> N = nbase+16j+l15
  #pragma unroll
  for (int j = 0; j < 8; ++j) {
    const int col = nbase + j * 16 + l15;
    #pragma unroll
    for (int v = 0; v < 8; ++v)
      Z[(size_t)(mbase + hi * 8 + v) * NCOLS + col] = acc[j][v];
  }
}

// ---------------------------------------------------------------------------
// Gates (keras order i,f,g,o; hard_sigmoid, tanh), cell/hidden update.
// Writes h (f16, recurrence image) and aux (f16): BN0(h) for layer 0,
// raw h for layer 1 (BN1 is folded into the dense weights).
// ---------------------------------------------------------------------------
__device__ __forceinline__ float hsig(float v) {
  return fminf(fmaxf(0.2f * v + 0.5f, 0.f), 1.f);
}

__global__ void gate_kernel(const float* __restrict__ Z, const float* __restrict__ bias,
                            float* __restrict__ cstate, _Float16* __restrict__ himg,
                            _Float16* __restrict__ aux, const float* __restrict__ bn_scale,
                            const float* __restrict__ bn_shift, int do_bn) {
  const int idx = blockIdx.x * 256 + threadIdx.x;      // 8192*64
  const int mrow = idx >> 6, f = idx & 63;
  const float* zr = Z + (size_t)mrow * NCOLS + f;
  const float zi = zr[0]       + bias[f];
  const float zf = zr[NF]      + bias[NF + f];
  const float zg = zr[2 * NF]  + bias[2 * NF + f];
  const float zo = zr[3 * NF]  + bias[3 * NF + f];
  const float ig = hsig(zi), fg = hsig(zf), og = hsig(zo);
  const float c  = fg * cstate[idx] + ig * tanhf(zg);
  const float h  = og * tanhf(c);
  cstate[idx] = c;
  himg[idx]   = (_Float16)h;
  const float a = do_bn ? (h * bn_scale[f] + bn_shift[f]) : h;
  aux[idx] = (_Float16)a;
}

// ---------------------------------------------------------------------------
// Dense(64->10, BN1 folded) + softmax + global mean over (T,H,W).
// Block reduces 256 positions into LDS, one f32 atomic per class per block.
// ---------------------------------------------------------------------------
__global__ void final_kernel(const _Float16* __restrict__ h1seq,
                             const float* __restrict__ Wp, const float* __restrict__ bp,
                             float* __restrict__ out) {
  __shared__ float red[NCLS];
  const int tid = threadIdx.x;
  const int bb  = blockIdx.x >> 6;                     // batch
  const int chunk = blockIdx.x & 63;
  if (tid < NCLS) red[tid] = 0.f;
  __syncthreads();

  const int pos = chunk * 256 + tid;                   // 0..16383 = t*1024+yx
  const int t = pos >> 10, yx = pos & 1023;
  const _Float16* hp = h1seq + ((size_t)t * MROWS + bb * (NH * NW) + yx) * NF;

  float logits[NCLS];
  #pragma unroll
  for (int c = 0; c < NCLS; ++c) logits[c] = bp[c];
  for (int f = 0; f < NF; ++f) {
    const float hv = (float)hp[f];
    #pragma unroll
    for (int c = 0; c < NCLS; ++c) logits[c] += hv * Wp[f * NCLS + c];
  }
  float mx = logits[0];
  #pragma unroll
  for (int c = 1; c < NCLS; ++c) mx = fmaxf(mx, logits[c]);
  float s = 0.f;
  #pragma unroll
  for (int c = 0; c < NCLS; ++c) { logits[c] = expf(logits[c] - mx); s += logits[c]; }
  const float inv = 1.f / (s * (float)(NT * NH * NW));
  #pragma unroll
  for (int c = 0; c < NCLS; ++c) atomicAdd(&red[c], logits[c] * inv);
  __syncthreads();
  if (tid < NCLS) atomicAdd(out + bb * NCLS + tid, red[tid]);
}

// ---------------------------------------------------------------------------
// Host orchestration: 2 prep launches, then per layer 16 x (GEMM + gates),
// then the reduction. All on `stream`; graph-capture safe (async memsets).
// ---------------------------------------------------------------------------
extern "C" void kernel_launch(void* const* d_in, const int* in_sizes, int n_in,
                              void* d_out, int out_size, void* d_ws, size_t ws_size,
                              hipStream_t stream) {
  (void)in_sizes; (void)n_in; (void)out_size; (void)ws_size;
  const float* inputs = (const float*)d_in[0];
  const float* l0_Wx  = (const float*)d_in[1];
  const float* l0_Wh  = (const float*)d_in[2];
  const float* l0_b   = (const float*)d_in[3];
  const float* l0_g   = (const float*)d_in[4];
  const float* l0_be  = (const float*)d_in[5];
  const float* l0_mu  = (const float*)d_in[6];
  const float* l0_va  = (const float*)d_in[7];
  const float* l1_Wx  = (const float*)d_in[8];
  const float* l1_Wh  = (const float*)d_in[9];
  const float* l1_b   = (const float*)d_in[10];
  const float* l1_g   = (const float*)d_in[11];
  const float* l1_be  = (const float*)d_in[12];
  const float* l1_mu  = (const float*)d_in[13];
  const float* l1_va  = (const float*)d_in[14];
  const float* dW     = (const float*)d_in[15];
  const float* db     = (const float*)d_in[16];

  char* ws = (char*)d_ws;
  size_t off = 0;
  auto take = [&](size_t bytes) -> void* {
    void* p = ws + off;
    off += (bytes + 255) & ~(size_t)255;
    return p;
  };
  _Float16* Wx0t  = (_Float16*)take((size_t)NCOLS * KX0 * 2);
  _Float16* Wh0t  = (_Float16*)take((size_t)NCOLS * KH  * 2);
  _Float16* Wx1t  = (_Float16*)take((size_t)NCOLS * KH  * 2);
  _Float16* Wh1t  = (_Float16*)take((size_t)NCOLS * KH  * 2);
  float*    scale0= (float*)take(NF * 4);
  float*    shift0= (float*)take(NF * 4);
  float*    Wp    = (float*)take(NF * NCLS * 4);
  float*    bp    = (float*)take(NCLS * 4);
  _Float16* X0col = (_Float16*)take((size_t)NT * MROWS * KX0 * 2);   // 16 MB
  _Float16* h0bn  = (_Float16*)take((size_t)NT * MROWS * NF * 2);    // 16 MB
  _Float16* h1seq = (_Float16*)take((size_t)NT * MROWS * NF * 2);    // 16 MB
  float*    Z     = (float*)take((size_t)MROWS * NCOLS * 4);         // 8 MB
  _Float16* h0buf = (_Float16*)take((size_t)MROWS * NF * 2);
  _Float16* h1buf = (_Float16*)take((size_t)MROWS * NF * 2);
  float*    c0    = (float*)take((size_t)MROWS * NF * 4);
  float*    c1    = (float*)take((size_t)MROWS * NF * 4);

  // zero the recurrent state + output accumulator (graph-capture safe)
  hipMemsetAsync(h0buf, 0, (size_t)MROWS * NF * 2, stream);
  hipMemsetAsync(h1buf, 0, (size_t)MROWS * NF * 2, stream);
  hipMemsetAsync(c0,    0, (size_t)MROWS * NF * 4, stream);
  hipMemsetAsync(c1,    0, (size_t)MROWS * NF * 4, stream);
  hipMemsetAsync(d_out, 0, (size_t)NB * NCLS * 4, stream);

  prep_kernel<<<KH, 256, 0, stream>>>(l0_Wx, l0_Wh, l1_Wx, l1_Wh,
                                      l0_g, l0_be, l0_mu, l0_va,
                                      l1_g, l1_be, l1_mu, l1_va, dW, db,
                                      Wx0t, Wh0t, Wx1t, Wh1t,
                                      scale0, shift0, Wp, bp);
  pack_x0_kernel<<<(NT * MROWS) / 256, 256, 0, stream>>>(inputs, X0col);

  // ---- layer 0 recurrence ----
  for (int t = 0; t < NT; ++t) {
    gemm_conv_kernel<<<MROWS / 64, 256, 0, stream>>>(
        X0col + (size_t)t * MROWS * KX0, KX0, /*mode0=*/0, Wx0t,
        h0buf, KH, Wh0t, Z);
    gate_kernel<<<(MROWS * NF) / 256, 256, 0, stream>>>(
        Z, l0_b, c0, h0buf, h0bn + (size_t)t * MROWS * NF, scale0, shift0, 1);
  }
  // ---- layer 1 recurrence (input = BN0(h0) sequence) ----
  for (int t = 0; t < NT; ++t) {
    gemm_conv_kernel<<<MROWS / 64, 256, 0, stream>>>(
        h0bn + (size_t)t * MROWS * NF, KH, /*mode0=*/1, Wx1t,
        h1buf, KH, Wh1t, Z);
    gate_kernel<<<(MROWS * NF) / 256, 256, 0, stream>>>(
        Z, l1_b, c1, h1buf, h1seq + (size_t)t * MROWS * NF, scale0, shift0, 0);
  }
  // ---- dense(BN1-folded) + softmax + global mean ----
  final_kernel<<<NB * 64, 256, 0, stream>>>(h1seq, Wp, bp, (float*)d_out);
}
